// MultiHeadGraphAttention_58402965291281
// MI455X (gfx1250) — matrix operand order repaired and checked by reading, not compiled, not run on hardware
//
#include <hip/hip_runtime.h>
#include <hip/hip_bf16.h>

// ---------------------------------------------------------------------------
// MultiHeadGraphAttention on MI455X (gfx1250, wave32)
//   N=100000 nodes, E=1600000 edges, DIM=128, HEADS=8, HEAD_DIM=16
// Dense projections: v_wmma_f32_16x16x32_bf16 (bf16 operands, f32 accum),
// W staged once per block in LDS as bf16, 5 M-tiles per wave per B operand.
// Edge phase: f32, bandwidth-bound; Q/K/V arrays (51 MB each) are L2-resident.
// ---------------------------------------------------------------------------

typedef __bf16 bf16_t;
typedef __attribute__((ext_vector_type(16))) __bf16 v16bf;
typedef __attribute__((ext_vector_type(8)))  float  v8f;

#define DIM 128
#define HEADS 8
#define INV_SCALE 0.25f   // 1/sqrt(HEAD_DIM=16)
#define MTILES 5          // 16-row tiles per wave -> 80-row M-block per block

// ---- monotone float <-> uint key for atomicMax-based segment max -----------
__device__ __forceinline__ unsigned f2key(float f) {
    unsigned b = __float_as_uint(f);
    return (b & 0x80000000u) ? ~b : (b | 0x80000000u);
}
__device__ __forceinline__ float key2f(unsigned k) {
    unsigned b = (k & 0x80000000u) ? (k & 0x7fffffffu) : ~k;
    return __uint_as_float(b);
}

// ---------------------------------------------------------------------------
// GEMM: Y[nrows x 128] = X[nrows x 128] @ W[128 x 128] + bias[128]
// Block = 256 threads = 8 waves. Wave w owns column tile n0 = 16*w and
// MTILES row tiles at m0 = 80*blockIdx.x. K=128 consumed as 4 chunks of 32
// via v_wmma_f32_16x16x32_bf16 (operand k order used by this code):
//   A 16x32 : lane-half 0 holds K 0..15, half 1 holds K 16..31
//   B 32x16 : lane-half 0 holds K 0..15, half 1 holds K 16..31, col = lane&15
//   D       : VGPR v -> row v + 8*(lane>=16), col = lane&15
// W is converted to bf16 and staged in LDS once per block; each lane's B
// operand per chunk is one contiguous 32-byte LDS read.
// ---------------------------------------------------------------------------
__global__ __launch_bounds__(256) void gemm128_wmma(
    const float* __restrict__ X, const float* __restrict__ W,
    const float* __restrict__ bias, float* __restrict__ Y, int nrows) {
    // LDS W layout: lw[(((kt*2 + half)*128 + col) * 16) + j]  (bf16), 32 KB
    __shared__ __attribute__((aligned(32))) __bf16 lw[4 * 2 * 128 * 16];

    // Cooperative, coalesced fill: idx sweeps W linearly (k major, col minor)
    for (int idx = threadIdx.x; idx < DIM * DIM; idx += 256) {
        const int k = idx >> 7, col = idx & 127;
        const int kt = k >> 5, hf = (k >> 4) & 1, j = k & 15;
        lw[(((kt * 2 + hf) * 128 + col) << 4) + j] = (bf16_t)W[idx];
    }
    __syncthreads();

    const int wave = threadIdx.x >> 5;
    const int lane = threadIdx.x & 31;
    const int half = lane >> 4;     // 0: lanes 0-15, 1: lanes 16-31
    const int l    = lane & 15;
    const int n0   = wave * 16;
    const int m0   = blockIdx.x * (16 * MTILES);
    const float bcol = bias[n0 + l];

    v8f c[MTILES];
#pragma unroll
    for (int t = 0; t < MTILES; ++t)
#pragma unroll
        for (int i = 0; i < 8; ++i) c[t][i] = bcol;   // accumulate on bias

#pragma unroll
    for (int kt = 0; kt < 4; ++kt) {
        // B operand: 16 bf16 = one 32-byte aligned LDS read
        const v16bf b = *(const v16bf*)(lw + (((kt * 2 + half) * 128 + (n0 + l)) << 4));
        const int ka = kt * 32 + half * 16;
#pragma unroll
        for (int t = 0; t < MTILES; ++t) {
            int row = m0 + t * 16 + l;
            if (row >= nrows) row = nrows - 1;        // clamp (store is guarded)
            const float* arow = X + (size_t)row * DIM;
            v16bf a;
#pragma unroll
            for (int j = 0; j < 8; ++j) a[j]     = (bf16_t)arow[ka + j];
#pragma unroll
            for (int j = 0; j < 8; ++j) a[8 + j] = (bf16_t)arow[ka + 8 + j];
            c[t] = __builtin_amdgcn_wmma_f32_16x16x32_bf16(
                       /*neg_a=*/false, a, /*neg_b=*/false, b,
                       /*c_mod=*/(short)0, c[t],
                       /*reuse_a=*/false, /*reuse_b=*/false);
        }
    }

#pragma unroll
    for (int t = 0; t < MTILES; ++t)
#pragma unroll
        for (int v = 0; v < 8; ++v) {
            const int row = m0 + t * 16 + v + 8 * half;
            if (row < nrows) Y[(size_t)row * DIM + (n0 + l)] = c[t][v];
        }
}

// ---------------------------------------------------------------------------
// Workspace init (must run each call: agg/denom/segmax are accumulated into)
// ---------------------------------------------------------------------------
__global__ void init_ws_k(unsigned* __restrict__ segmax, float* __restrict__ denom,
                          float* __restrict__ agg, int n8, int n128) {
    int i = blockIdx.x * 256 + threadIdx.x;
    if (i < n8) { segmax[i] = 0u; denom[i] = 0.0f; }
    if (i < n128) agg[i] = 0.0f;
}

// ---------------------------------------------------------------------------
// Pass A: per-edge scores (E x 8) + segment max (atomicMax on monotone keys).
// One wave per edge; lane owns 4 of the 128 dims (head = lane/4).
// ---------------------------------------------------------------------------
__global__ __launch_bounds__(256) void edge_scores_k(
    const float* __restrict__ Qp, const float* __restrict__ Kp,
    const float* __restrict__ EA, const float* __restrict__ we,
    const float* __restrict__ be, const int* __restrict__ src,
    const int* __restrict__ dst, float* __restrict__ scores,
    unsigned* __restrict__ segmax, int E) {
    const int e = blockIdx.x * 8 + (threadIdx.x >> 5);
    if (e >= E) return;
    const int lane = threadIdx.x & 31;
    const int s = src[e], d = dst[e];

    // keep the HBM edge_attr stream deep (global_prefetch_b8, speculative)
    __builtin_prefetch(EA + (size_t)(e + 32) * DIM + lane * 4, 0, 1);

    // qk partial for head h = lane/4 (16 dims per head = 4 lanes x float4)
    const float4 q  = *(const float4*)(Qp + (size_t)d * DIM + lane * 4);
    const float4 kk = *(const float4*)(Kp + (size_t)s * DIM + lane * 4);
    float qk = q.x * kk.x + q.y * kk.y + q.z * kk.z + q.w * kk.w;
    qk += __shfl_xor(qk, 1, 32);
    qk += __shfl_xor(qk, 2, 32);          // lanes 4h..4h+3 all hold head-h dot

    // edge_attr @ we partials: lane covers k = lane*4 .. lane*4+3, all 8 heads
    const float4 ea = *(const float4*)(EA + (size_t)e * DIM + lane * 4);
    float p[HEADS];
#pragma unroll
    for (int h = 0; h < HEADS; ++h) {
        const float* wr = we + (size_t)(lane * 4) * HEADS + h;   // we[k][h], 128x8
        p[h] = ea.x * wr[0] + ea.y * wr[8] + ea.z * wr[16] + ea.w * wr[24];
    }
#pragma unroll
    for (int h = 0; h < HEADS; ++h) {
#pragma unroll
        for (int off = 16; off >= 1; off >>= 1) p[h] += __shfl_xor(p[h], off, 32);
    }

    // lane h (h<8) finalizes head h; fetch its qk from lane 4h (all lanes exec shfl)
    const float qkh = __shfl(qk, (lane & 7) * 4, 32);
    if (lane < HEADS) {
        const float sc = qkh * INV_SCALE + p[lane] + be[lane];
        scores[(size_t)e * HEADS + lane] = sc;
        atomicMax(segmax + (size_t)d * HEADS + lane, f2key(sc));
    }
}

// ---------------------------------------------------------------------------
// Pass B: denom[d][h] += exp(score - segmax[d][h]);  one thread per (e,h)
// ---------------------------------------------------------------------------
__global__ __launch_bounds__(256) void edge_denom_k(
    const float* __restrict__ scores, const unsigned* __restrict__ segmax,
    const int* __restrict__ dst, float* __restrict__ denom, int EH) {
    int i = blockIdx.x * 256 + threadIdx.x;
    if (i >= EH) return;
    const int e = i >> 3, h = i & 7;
    const int d = dst[e];
    const float m = key2f(segmax[(size_t)d * HEADS + h]);
    atomicAdd(denom + (size_t)d * HEADS + h, __expf(scores[i] - m));
}

// ---------------------------------------------------------------------------
// Pass C: agg[d] += V[s] * attn;  one wave per edge, 4 atomicAdds per lane
// ---------------------------------------------------------------------------
__global__ __launch_bounds__(256) void edge_scatter_k(
    const float* __restrict__ Vp, const float* __restrict__ scores,
    const unsigned* __restrict__ segmax, const float* __restrict__ denom,
    const int* __restrict__ src, const int* __restrict__ dst,
    float* __restrict__ agg, int E) {
    const int e = blockIdx.x * 8 + (threadIdx.x >> 5);
    if (e >= E) return;
    const int lane = threadIdx.x & 31;
    const int h = lane >> 2;
    const int s = src[e], d = dst[e];

    const float m   = key2f(segmax[(size_t)d * HEADS + h]);
    const float den = denom[(size_t)d * HEADS + h];
    const float a   = __expf(scores[(size_t)e * HEADS + h] - m) / den;

    const float4 v4 = *(const float4*)(Vp + (size_t)s * DIM + lane * 4);
    float* outp = agg + (size_t)d * DIM + lane * 4;
    atomicAdd(outp + 0, v4.x * a);
    atomicAdd(outp + 1, v4.y * a);
    atomicAdd(outp + 2, v4.z * a);
    atomicAdd(outp + 3, v4.w * a);
}

// ---------------------------------------------------------------------------
// Host launcher. Input order: query,key,value,edge_attr,wq,bq,wk,bk,wv,bv,
//                             wo,bo,we,be,edge_index
// ---------------------------------------------------------------------------
extern "C" void kernel_launch(void* const* d_in, const int* in_sizes, int n_in,
                              void* d_out, int out_size, void* d_ws, size_t ws_size,
                              hipStream_t stream) {
    const float* query     = (const float*)d_in[0];
    const float* key_      = (const float*)d_in[1];
    const float* value     = (const float*)d_in[2];
    const float* edge_attr = (const float*)d_in[3];
    const float* wq = (const float*)d_in[4];  const float* bq = (const float*)d_in[5];
    const float* wk = (const float*)d_in[6];  const float* bk = (const float*)d_in[7];
    const float* wv = (const float*)d_in[8];  const float* bv = (const float*)d_in[9];
    const float* wo = (const float*)d_in[10]; const float* bo = (const float*)d_in[11];
    const float* we = (const float*)d_in[12]; const float* be = (const float*)d_in[13];
    const int*   ei = (const int*)d_in[14];

    const int N = in_sizes[0] / DIM;          // 100000
    const int E = in_sizes[3] / DIM;          // 1600000
    const int* src = ei;                      // edge_index[0]
    const int* dst = ei + E;                  // edge_index[1]

    // workspace layout (floats): Qp | Kp | Vp | agg | segmax(u32) | denom | scores
    float*    Qp     = (float*)d_ws;
    float*    Kp     = Qp + (size_t)N * DIM;
    float*    Vp     = Kp + (size_t)N * DIM;
    float*    agg    = Vp + (size_t)N * DIM;
    unsigned* segmax = (unsigned*)(agg + (size_t)N * DIM);
    float*    denom  = (float*)(segmax + (size_t)N * HEADS);
    float*    scores = denom + (size_t)N * HEADS;

    const int n128 = N * DIM;
    init_ws_k<<<(n128 + 255) / 256, 256, 0, stream>>>(segmax, denom, agg, N * HEADS, n128);

    const dim3 gblk((N + 16 * MTILES - 1) / (16 * MTILES));
    gemm128_wmma<<<gblk, 256, 0, stream>>>(query, wq, bq, Qp, N);
    gemm128_wmma<<<gblk, 256, 0, stream>>>(key_,  wk, bk, Kp, N);
    gemm128_wmma<<<gblk, 256, 0, stream>>>(value, wv, bv, Vp, N);

    const int eblk = (E + 7) / 8;
    edge_scores_k <<<eblk, 256, 0, stream>>>(Qp, Kp, edge_attr, we, be, src, dst,
                                             scores, segmax, E);
    edge_denom_k  <<<(E * HEADS + 255) / 256, 256, 0, stream>>>(scores, segmax, dst,
                                                                denom, E * HEADS);
    edge_scatter_k<<<eblk, 256, 0, stream>>>(Vp, scores, segmax, denom, src, dst,
                                             agg, E);

    gemm128_wmma<<<gblk, 256, 0, stream>>>(agg, wo, bo, (float*)d_out, N);
}
